// MultiHeadSelfAttention_1563368096585
// MI455X (gfx1250) — compile-verified, matrix operand order checked by reading, not run on hardware
//
#include <hip/hip_runtime.h>
#include <hip/hip_bf16.h>

// ---------------------------------------------------------------------------
// MHA on gfx1250: fp32 inputs -> bf16 WMMA (v_wmma_f32_16x16x32_bf16) for all
// matmuls. Attention is fused flash-style with TRANSPOSED score/output tiles
// so P flows from WMMA accumulators straight into the next WMMA's B operand
// (C-layout of S^T == B-layout of P) -- no LDS staging, no shuffle chains.
// Each attention wave owns 32 queries (2 col-tiles) to double WMMA/byte;
// K-fragment loads are sequenced and the kernel uses launch_bounds(128,1)
// to keep the live set inside the 256 directly-addressable VGPRs (no spills).
// Causal mask handled by loop splitting: only the final k-block is masked.
// 1/sqrt(Dh) is folded into Wq at weight-conversion time (exact: 2^-3).
// ---------------------------------------------------------------------------

typedef __attribute__((ext_vector_type(16))) __bf16 v16bf;
typedef __attribute__((ext_vector_type(8)))  __bf16 v8bf;
typedef __attribute__((ext_vector_type(8)))  float  v8f;

#define D_MODEL  1024
#define NUM_H    16
#define HEAD_D   64
#define SEQ      2048
#define BATCH    2
#define M_TOT    (BATCH * SEQ)        // 4096
#define LDQKV    (3 * D_MODEL)        // 3072

#define WMMA_BF16(A_, B_, C_) \
  __builtin_amdgcn_wmma_f32_16x16x32_bf16(false, (A_), false, (B_), (short)0, (C_), false, false)

union FragU { v16bf v; v8bf h[2]; };

// Load one 16x32 bf16 WMMA operand fragment (A from row-major MxK, or B from
// "Bt" = N x K row-major). ISA layout: lane l -> row l%16; element e holds
// K = 16*(e/8) + 8*(l/16) + e%8  => two contiguous 8-element (16B) chunks.
__device__ __forceinline__ v16bf load_frag(const __bf16* base, int ld, int k0) {
  int lane = threadIdx.x & 31;
  int r = lane & 15, half = lane >> 4;
  const __bf16* p = base + (size_t)r * ld + k0 + half * 8;
  FragU f;
  f.h[0] = *(const v8bf*)(p);
  f.h[1] = *(const v8bf*)(p + 16);
  return f.v;
}

// Exchange value with lane^16 (the other 16-lane half) -- pure VALU on gfx1250.
__device__ __forceinline__ float xor16(float x) {
#if __has_builtin(__builtin_amdgcn_permlanex16)
  int i = __builtin_amdgcn_permlanex16(__float_as_int(x), __float_as_int(x),
                                       0x76543210, 0xfedcba98, false, false);
  return __int_as_float(i);
#else
  return __shfl_xor(x, 16);
#endif
}

// ---------------------------------------------------------------------------
// 1) fp32 -> bf16 elementwise convert (x)
// ---------------------------------------------------------------------------
__global__ void convert_f32_bf16(const float* __restrict__ in, __bf16* __restrict__ out, int n) {
  int i = (blockIdx.x * blockDim.x + threadIdx.x) * 4;
  if (i + 3 < n) {
    float4 f = *(const float4*)(in + i);
    out[i + 0] = (__bf16)f.x;
    out[i + 1] = (__bf16)f.y;
    out[i + 2] = (__bf16)f.z;
    out[i + 3] = (__bf16)f.w;
  }
}

// ---------------------------------------------------------------------------
// 2) fp32 [1024x1024] -> bf16 transposed+scaled (Wt[n][k] = W[k][n] * scale)
// ---------------------------------------------------------------------------
__global__ void transpose_w(const float* __restrict__ W, __bf16* __restrict__ Wt, float scale) {
  __shared__ float tile[32][33];
  int n0 = blockIdx.x * 32, k0 = blockIdx.y * 32;
  for (int i = threadIdx.y; i < 32; i += 8)
    tile[i][threadIdx.x] = W[(size_t)(k0 + i) * D_MODEL + n0 + threadIdx.x];
  __syncthreads();
  for (int i = threadIdx.y; i < 32; i += 8)
    Wt[(size_t)(n0 + i) * D_MODEL + k0 + threadIdx.x] = (__bf16)(tile[threadIdx.x][i] * scale);
}

// ---------------------------------------------------------------------------
// 3) WMMA GEMM: C[M x N] = A[M x K] * Bt[N x K]^T, bf16 in, f32 accumulate.
//    256 threads = 8 waves; each wave a 32x64 tile (2x4 WMMA, 8 WMMA per
//    K-step off 6 fragment loads); block tile 64 x 256.
// ---------------------------------------------------------------------------
template <bool OUT_BF16>
__global__ void gemm_bf16_wmma(const __bf16* __restrict__ A, const __bf16* __restrict__ Bt,
                               void* __restrict__ Cout, int K, int lda, int ldb, int ldc) {
  int wave = threadIdx.x >> 5;
  int lane = threadIdx.x & 31;
  int m0 = blockIdx.y * 64 + (wave & 1) * 32;
  int n0 = blockIdx.x * 256 + (wave >> 1) * 64;

  v8f acc[2][4];
#pragma unroll
  for (int i = 0; i < 2; ++i)
#pragma unroll
    for (int j = 0; j < 4; ++j) acc[i][j] = (v8f){0.f, 0.f, 0.f, 0.f, 0.f, 0.f, 0.f, 0.f};

  const __bf16* Ab = A + (size_t)m0 * lda;
  const __bf16* Bb = Bt + (size_t)n0 * ldb;
  int r = lane & 15, half = lane >> 4;

  for (int k0 = 0; k0 < K; k0 += 32) {
    if (k0 + 32 < K) {  // global_prefetch_b8 of next K-slice
      __builtin_prefetch(Ab + (size_t)r * lda + k0 + 32, 0, 0);
      __builtin_prefetch(Bb + (size_t)r * ldb + k0 + 32, 0, 0);
    }
    v16bf a0 = load_frag(Ab, lda, k0);
    v16bf a1 = load_frag(Ab + (size_t)16 * lda, lda, k0);
#pragma unroll
    for (int j = 0; j < 4; ++j) {
      v16bf b = load_frag(Bb + (size_t)(j * 16) * ldb, ldb, k0);
      acc[0][j] = WMMA_BF16(a0, b, acc[0][j]);
      acc[1][j] = WMMA_BF16(a1, b, acc[1][j]);
    }
  }

#pragma unroll
  for (int i = 0; i < 2; ++i)
#pragma unroll
    for (int j = 0; j < 4; ++j)
#pragma unroll
      for (int v = 0; v < 8; ++v) {
        int row = m0 + i * 16 + v + 8 * half;
        int col = n0 + j * 16 + r;
        if (OUT_BF16)
          ((__bf16*)Cout)[(size_t)row * ldc + col] = (__bf16)acc[i][j][v];
        else
          ((float*)Cout)[(size_t)row * ldc + col] = acc[i][j][v];
      }
}

// ---------------------------------------------------------------------------
// 4) Build V^T per head: vt[b][h][d][s] = qkv[b*S+s][2048 + h*64 + d]
// ---------------------------------------------------------------------------
__global__ void transpose_v(const __bf16* __restrict__ qkv, __bf16* __restrict__ vt) {
  int idx = blockIdx.x * blockDim.x + threadIdx.x;  // [b][h][d][s] flat, 4M total
  int s = idx & (SEQ - 1);
  int d = (idx >> 11) & (HEAD_D - 1);
  int h = (idx >> 17) & (NUM_H - 1);
  int b = idx >> 21;
  vt[idx] = qkv[((size_t)(b * SEQ + s)) * LDQKV + 2 * D_MODEL + h * HEAD_D + d];
}

// ---------------------------------------------------------------------------
// 5) Fused causal flash attention, transposed-tile formulation.
//    128 threads = 4 waves; each wave owns one (b, h, 32-query block):
//      S^T = K * Q^T  (A = K rows, B = Q rows)  -> lane owns one query column
//      O^T = V^T * P  (A = V^T rows, B = P)     -> P straight from registers
//    Two query col-tiles (A/B) share every K/V fragment: 16 WMMA / 8 loads.
//    K fragments are loaded in two sequenced pairs to cap register liveness.
//    Loop split: nfull = q0/32 unmasked blocks, then exactly 1 masked block.
// ---------------------------------------------------------------------------

// One 32-key block step; MASKED selects causal-mask code at compile time.
#define ATTN_KBLOCK(KT, MASKED)                                                   \
  {                                                                               \
    const int kt = (KT);                                                          \
    /* key subtile 0: load 2 frags, 4 WMMAs, then REUSE the registers */          \
    v16bf kfa = load_frag(Kb + (size_t)kt * LDQKV, LDQKV, 0);                     \
    v16bf kfb = load_frag(Kb + (size_t)kt * LDQKV, LDQKV, 32);                    \
    v8f zz = (v8f){0.f, 0.f, 0.f, 0.f, 0.f, 0.f, 0.f, 0.f};                       \
    v8f s0A = WMMA_BF16(kfa, qA0, zz); s0A = WMMA_BF16(kfb, qA1, s0A);            \
    v8f s0B = WMMA_BF16(kfa, qB0, zz); s0B = WMMA_BF16(kfb, qB1, s0B);            \
    /* key subtile 1 */                                                           \
    kfa = load_frag(Kb + (size_t)(kt + 16) * LDQKV, LDQKV, 0);                    \
    kfb = load_frag(Kb + (size_t)(kt + 16) * LDQKV, LDQKV, 32);                   \
    v8f s1A = WMMA_BF16(kfa, qA0, zz); s1A = WMMA_BF16(kfb, qA1, s1A);            \
    v8f s1B = WMMA_BF16(kfa, qB0, zz); s1B = WMMA_BF16(kfb, qB1, s1B);            \
    float mxA = NEG_INF, mxB = NEG_INF;                                           \
    _Pragma("unroll")                                                             \
    for (int v = 0; v < 8; ++v) {                                                 \
      if (MASKED) {                                                               \
        int key0 = kt + v + 8 * half, key1 = key0 + 16;                           \
        if (key0 > qqA) s0A[v] = NEG_INF;                                         \
        if (key1 > qqA) s1A[v] = NEG_INF;                                         \
        if (key0 > qqB) s0B[v] = NEG_INF;                                         \
        if (key1 > qqB) s1B[v] = NEG_INF;                                         \
      }                                                                           \
      mxA = fmaxf(mxA, fmaxf(s0A[v], s1A[v]));                                    \
      mxB = fmaxf(mxB, fmaxf(s0B[v], s1B[v]));                                    \
    }                                                                             \
    mxA = fmaxf(mxA, xor16(mxA));                                                 \
    mxB = fmaxf(mxB, xor16(mxB));                                                 \
    float mnA = fmaxf(mrunA, mxA), mnB = fmaxf(mrunB, mxB);                       \
    float alA = __expf(mrunA - mnA), alB = __expf(mrunB - mnB);                   \
    mrunA = mnA; mrunB = mnB;                                                     \
    v16bf pA, pB;                                                                 \
    float rsA = 0.f, rsB = 0.f;                                                   \
    _Pragma("unroll")                                                             \
    for (int v = 0; v < 8; ++v) {                                                 \
      float e0 = __expf(s0A[v] - mnA), e1 = __expf(s1A[v] - mnA);                 \
      rsA += e0 + e1; pA[v] = (__bf16)e0; pA[v + 8] = (__bf16)e1;                 \
      float f0 = __expf(s0B[v] - mnB), f1 = __expf(s1B[v] - mnB);                 \
      rsB += f0 + f1; pB[v] = (__bf16)f0; pB[v + 8] = (__bf16)f1;                 \
    }                                                                             \
    rsA += xor16(rsA); rsB += xor16(rsB);                                         \
    lrunA = lrunA * alA + rsA; lrunB = lrunB * alB + rsB;                         \
    _Pragma("unroll")                                                             \
    for (int n = 0; n < 4; ++n) {                                                 \
      v16bf vf = load_frag(Vh + (size_t)(n * 16) * SEQ, SEQ, kt);                 \
      otA[n] = otA[n] * alA; otA[n] = WMMA_BF16(vf, pA, otA[n]);                  \
      otB[n] = otB[n] * alB; otB[n] = WMMA_BF16(vf, pB, otB[n]);                  \
    }                                                                             \
  }

__global__ void __launch_bounds__(128, 1)
attn_kernel(const __bf16* __restrict__ qkv, const __bf16* __restrict__ vt,
            __bf16* __restrict__ attnb) {
  const float NEG_INF = -__builtin_inff();

  int wave = threadIdx.x >> 5, lane = threadIdx.x & 31;
  int gid = blockIdx.x * 4 + wave;           // 2048 total
  int qb = gid & 63;                         // S/32 = 64 query blocks
  int hh = (gid >> 6) & (NUM_H - 1);
  int bb = gid >> 10;
  int q0 = qb * 32;
  int r = lane & 15, half = lane >> 4;
  int qqA = q0 + r, qqB = q0 + 16 + r;       // this lane's two query indices

  const __bf16* Qb = qkv + ((size_t)(bb * SEQ + q0)) * LDQKV + hh * HEAD_D;
  const __bf16* Kb = qkv + ((size_t)(bb * SEQ)) * LDQKV + D_MODEL + hh * HEAD_D;
  const __bf16* Vh = vt + ((size_t)(bb * NUM_H + hh)) * HEAD_D * SEQ;  // [64][S]

  // Q as B-operand (lane = query column); Dh=64 -> two K=32 fragments per tile.
  v16bf qA0 = load_frag(Qb, LDQKV, 0);
  v16bf qA1 = load_frag(Qb, LDQKV, 32);
  v16bf qB0 = load_frag(Qb + (size_t)16 * LDQKV, LDQKV, 0);
  v16bf qB1 = load_frag(Qb + (size_t)16 * LDQKV, LDQKV, 32);

  v8f otA[4], otB[4];                         // O^T: 4 tiles of [16 d x 16 q] each
#pragma unroll
  for (int n = 0; n < 4; ++n) {
    otA[n] = (v8f){0.f, 0.f, 0.f, 0.f, 0.f, 0.f, 0.f, 0.f};
    otB[n] = otA[n];
  }
  float mrunA = NEG_INF, lrunA = 0.f;         // per-lane (= per-query) stats
  float mrunB = NEG_INF, lrunB = 0.f;

  int nfull = q0 >> 5;                        // blocks with all keys <= all queries
  for (int kb = 0; kb < nfull; ++kb) ATTN_KBLOCK(kb * 32, false)
  ATTN_KBLOCK(nfull * 32, true)               // the single diagonal block

  // normalize; O^T epilogue is contiguous per lane -> packed 16B stores
  float invA = 1.f / lrunA, invB = 1.f / lrunB;
#pragma unroll
  for (int n = 0; n < 4; ++n) {
    v8bf ovA, ovB;
#pragma unroll
    for (int v = 0; v < 8; ++v) {
      ovA[v] = (__bf16)(otA[n][v] * invA);
      ovB[v] = (__bf16)(otB[n][v] * invB);
    }
    size_t coloff = (size_t)hh * HEAD_D + n * 16 + 8 * half;
    *(v8bf*)(attnb + ((size_t)(bb * SEQ + qqA)) * D_MODEL + coloff) = ovA;
    *(v8bf*)(attnb + ((size_t)(bb * SEQ + qqB)) * D_MODEL + coloff) = ovB;
  }
}

// ---------------------------------------------------------------------------
extern "C" void kernel_launch(void* const* d_in, const int* in_sizes, int n_in,
                              void* d_out, int out_size, void* d_ws, size_t ws_size,
                              hipStream_t stream) {
  const float* x  = (const float*)d_in[0];
  const float* Wq = (const float*)d_in[1];
  const float* Wk = (const float*)d_in[2];
  const float* Wv = (const float*)d_in[3];
  const float* Wo = (const float*)d_in[4];
  float* out = (float*)d_out;

  // workspace layout (≈56 MB total)
  char* ws = (char*)d_ws;
  __bf16* xb    = (__bf16*)(ws);                          //  8 MB  x bf16 [4096x1024]
  __bf16* wqkvT = (__bf16*)(ws + ((size_t)8  << 20));     //  6 MB  [Wq^T;Wk^T;Wv^T] [3072x1024]
  __bf16* woT   = (__bf16*)(ws + ((size_t)14 << 20));     //  2 MB  Wo^T [1024x1024]
  __bf16* qkv   = (__bf16*)(ws + ((size_t)16 << 20));     // 24 MB  QKV [4096x3072]
  __bf16* vtb   = (__bf16*)(ws + ((size_t)40 << 20));     //  8 MB  V^T [B,H,64,S]
  __bf16* attnb = (__bf16*)(ws + ((size_t)48 << 20));     //  8 MB  attn out [4096x1024]

  // 1) x -> bf16
  convert_f32_bf16<<<M_TOT * D_MODEL / 1024, 256, 0, stream>>>(x, xb, M_TOT * D_MODEL);

  // 2) weights -> bf16 transposed; fold 1/sqrt(Dh)=0.125 into Wq (exact pow2)
  dim3 tb(32, 8), tg(32, 32);
  transpose_w<<<tg, tb, 0, stream>>>(Wq, wqkvT, 0.125f);
  transpose_w<<<tg, tb, 0, stream>>>(Wk, wqkvT + (size_t)D_MODEL * D_MODEL, 1.0f);
  transpose_w<<<tg, tb, 0, stream>>>(Wv, wqkvT + (size_t)2 * D_MODEL * D_MODEL, 1.0f);
  transpose_w<<<tg, tb, 0, stream>>>(Wo, woT, 1.0f);

  // 3) QKV = xb @ [Wq|Wk|Wv]  (M=4096, N=3072, K=1024)
  gemm_bf16_wmma<true><<<dim3(LDQKV / 256, M_TOT / 64), 256, 0, stream>>>(
      xb, wqkvT, qkv, D_MODEL, D_MODEL, D_MODEL, LDQKV);

  // 4) V -> V^T per head
  transpose_v<<<(BATCH * NUM_H * HEAD_D * SEQ) / 256, 256, 0, stream>>>(qkv, vtb);

  // 5) fused causal attention (2048 query-block waves, 4 waves/block)
  attn_kernel<<<(BATCH * NUM_H * (SEQ / 32)) / 4, 128, 0, stream>>>(qkv, vtb, attnb);

  // 6) out = attn @ Wo  (M=4096, N=1024, K=1024), fp32 output
  gemm_bf16_wmma<false><<<dim3(D_MODEL / 256, M_TOT / 64), 256, 0, stream>>>(
      attnb, woT, out, D_MODEL, D_MODEL, D_MODEL, D_MODEL);
}
